// SpatialAttention_16174846836977
// MI455X (gfx1250) — compile-verified, matrix operand order checked by reading, not run on hardware
//
#include <hip/hip_runtime.h>
#include <hip/hip_bf16.h>

// ---------------------------------------------------------------------------
// SpatialAttention for MI455X (gfx1250), flash-attention style:
//  - never materializes the 8x4096x4096 attention matrix (512MB -> 0)
//  - QK^T   : V_WMMA_F32_16X16X4_F32  (fp32 logits, K-dim = head dim = 8)
//  - softmax: online, per-lane (S^T layout => query = lane column)
//  - PV     : V_WMMA_F32_16X16X32_BF16 (K-dim = 4096 contraction)
//  - P^T in C/D layout packs directly into the bf16 B-operand layout
// ---------------------------------------------------------------------------

typedef __attribute__((ext_vector_type(2)))  float  v2f;
typedef __attribute__((ext_vector_type(8)))  float  v8f;
typedef __attribute__((ext_vector_type(16))) __bf16 v16bf;

#define NTOK 4096   // 16*16*16 tokens
#define CCH  64     // channels
#define HEADS 8
#define HDIM  8

__device__ __forceinline__ unsigned short f32_bf16(float f) {
  unsigned u = __float_as_uint(f);
  u += 0x7FFFu + ((u >> 16) & 1u);   // round-to-nearest-even
  return (unsigned short)(u >> 16);
}

// ---------------------------------------------------------------------------
// Kernel 1: qkv[192][4096] = w_qkv(192x64) @ x(64x4096) + b_qkv
//           also emits bf16 V in [h][d_pad16][m] layout (pad rows zeroed)
// grid (256, 16) x 256 threads: blockIdx.x = output row (192 real + 64 pad)
// ---------------------------------------------------------------------------
__global__ __launch_bounds__(256) void qkv_kernel(
    const float* __restrict__ x, const float* __restrict__ w,
    const float* __restrict__ b, float* __restrict__ qkv,
    unsigned short* __restrict__ vbf) {
  const int co = blockIdx.x;
  const int n  = blockIdx.y * 256 + threadIdx.x;
  if (co >= 3 * CCH) {                 // zero-fill V pad rows d=8..15
    const int p = co - 3 * CCH;        // 0..63
    const int h = p >> 3, d = (p & 7) + 8;
    vbf[((h * 16) + d) * NTOK + n] = 0;
    return;
  }
  const float* wr = w + co * CCH;
  float acc = b[co];
#pragma unroll
  for (int ci = 0; ci < CCH; ++ci)
    acc = fmaf(wr[ci], x[ci * NTOK + n], acc);
  qkv[co * NTOK + n] = acc;
  if (co >= 2 * CCH) {                 // V -> bf16 copy
    const int c = co - 2 * CCH;
    const int h = c >> 3, d = c & 7;
    vbf[((h * 16) + d) * NTOK + n] = f32_bf16(acc);
  }
}

// ---------------------------------------------------------------------------
// Kernel 2: flash attention. One wave = one 16-query tile of one head.
// grid(256) x 256 threads (8 waves/block, all waves of a block share a head).
// ---------------------------------------------------------------------------
__global__ __launch_bounds__(256) void attn_kernel(
    const float* __restrict__ qkv, const unsigned short* __restrict__ vbf,
    float* __restrict__ attn_out) {
  const int lane = threadIdx.x & 31;
  const int wave = threadIdx.x >> 5;
  const int h    = blockIdx.x >> 5;
  const int tile = (blockIdx.x & 31) * 8 + wave;   // 0..255
  const int n0   = tile * 16;
  const int half = lane >> 4;                      // 0 | 1
  const int l16  = lane & 15;
  const float scale = 0.35355339059327373f;        // 1/sqrt(8)

  const float*          qb = qkv + (h * HDIM) * NTOK;            // q[d][n]
  const float*          kb = qkv + (CCH + h * HDIM) * NTOK;      // k[d][m]
  const unsigned short* vb = vbf + (h * 16) * NTOK;              // v[d16][m]

  // Q as fp32 B-operand (4x16): lane L, vgpr r -> q[K = r + 2*half][n0+l16]
  const int dq = 2 * half;
  v2f qB0, qB1;   // d = 0..3 and d = 4..7  (scale folded in)
  qB0.x = qb[(dq + 0) * NTOK + n0 + l16] * scale;
  qB0.y = qb[(dq + 1) * NTOK + n0 + l16] * scale;
  qB1.x = qb[(dq + 4) * NTOK + n0 + l16] * scale;
  qB1.y = qb[(dq + 5) * NTOK + n0 + l16] * scale;

  v8f   acc = {};                  // out^T accumulator (rows d, cols n)
  float run_max = -INFINITY;
  float run_sum = 0.0f;

  for (int m0 = 0; m0 < NTOK; m0 += 32) {
    // ---- K as fp32 A-operand (16x4): lane L, vgpr r -> k[r+2*half][m]
    const float* k0 = kb + m0 + l16;        // key tile 0 (m0..m0+15)
    const float* k1 = k0 + 16;              // key tile 1 (m0+16..m0+31)
    v2f kA00, kA01, kA10, kA11;
    kA00.x = k0[(dq + 0) * NTOK]; kA00.y = k0[(dq + 1) * NTOK];
    kA01.x = k0[(dq + 4) * NTOK]; kA01.y = k0[(dq + 5) * NTOK];
    kA10.x = k1[(dq + 0) * NTOK]; kA10.y = k1[(dq + 1) * NTOK];
    kA11.x = k1[(dq + 4) * NTOK]; kA11.y = k1[(dq + 5) * NTOK];

    // prefetch next chunk of V (gfx1250 global_prefetch path)
    __builtin_prefetch(vb + (size_t)l16 * NTOK + m0 + 32, 0, 1);

    // ---- S^T = (K-tile) x (Q-tile): fp32 WMMA, K-dim 4+4
    v8f s0 = {}, s1 = {};
    s0 = __builtin_amdgcn_wmma_f32_16x16x4_f32(false, kA00, false, qB0,
                                               (short)0, s0, false, false);
    s0 = __builtin_amdgcn_wmma_f32_16x16x4_f32(false, kA01, false, qB1,
                                               (short)0, s0, false, false);
    s1 = __builtin_amdgcn_wmma_f32_16x16x4_f32(false, kA10, false, qB0,
                                               (short)0, s1, false, false);
    s1 = __builtin_amdgcn_wmma_f32_16x16x4_f32(false, kA11, false, qB1,
                                               (short)0, s1, false, false);

    // ---- online softmax over this 32-key chunk (per lane = per query col)
    float cmax = s0[0];
#pragma unroll
    for (int r = 0; r < 8; ++r) {
      cmax = fmaxf(cmax, s0[r]);
      cmax = fmaxf(cmax, s1[r]);
    }
    cmax = fmaxf(cmax, __shfl_xor(cmax, 16, 32));      // join half-waves
    const float nmax = fmaxf(run_max, cmax);
    const float corr = __expf(run_max - nmax);
    run_max = nmax;

    float csum = 0.0f;
    v16bf pB;                       // P^T packed straight into B-operand
#pragma unroll
    for (int r = 0; r < 8; ++r) {
      const float p0 = __expf(s0[r] - nmax);
      const float p1 = __expf(s1[r] - nmax);
      csum += p0 + p1;
      pB[r]     = (__bf16)p0;       // K =  r + 8*half   (tile 0)
      pB[8 + r] = (__bf16)p1;       // K = 16 + r + 8*half (tile 1)
    }
    csum += __shfl_xor(csum, 16, 32);
    run_sum = run_sum * corr + csum;
#pragma unroll
    for (int r = 0; r < 8; ++r) acc[r] *= corr;

    // ---- V as bf16 A-operand: lane L (row d=l16), two 16B contiguous loads
    union { v16bf v; uint4 u[2]; } vA;
    const uint4* vp =
        (const uint4*)(vb + (size_t)l16 * NTOK + m0 + half * 8);
    vA.u[0] = vp[0];                // K =  0..7  (+8*half)
    vA.u[1] = vp[2];                // K = 16..23 (+8*half)

    acc = __builtin_amdgcn_wmma_f32_16x16x32_bf16(false, vA.v, false, pB,
                                                  (short)0, acc, false, false);
  }

  // ---- normalize + store: lanes 0..15 hold rows d=0..7 of out^T
  if (half == 0) {
    const float inv = 1.0f / run_sum;
    float* o = attn_out + (h * HDIM) * NTOK + n0 + l16;
#pragma unroll
    for (int r = 0; r < 8; ++r) o[r * NTOK] = acc[r] * inv;
  }
}

// ---------------------------------------------------------------------------
// Kernel 3: out = x + w_proj(64x64) @ attn_out + b_proj
// ---------------------------------------------------------------------------
__global__ __launch_bounds__(256) void proj_kernel(
    const float* __restrict__ x, const float* __restrict__ w,
    const float* __restrict__ b, const float* __restrict__ ao,
    float* __restrict__ out) {
  const int co = blockIdx.x;
  const int n  = blockIdx.y * 256 + threadIdx.x;
  const float* wr = w + co * CCH;
  float acc = b[co];
#pragma unroll
  for (int ci = 0; ci < CCH; ++ci)
    acc = fmaf(wr[ci], ao[ci * NTOK + n], acc);
  out[co * NTOK + n] = x[co * NTOK + n] + acc;
}

// ---------------------------------------------------------------------------
extern "C" void kernel_launch(void* const* d_in, const int* in_sizes, int n_in,
                              void* d_out, int out_size, void* d_ws,
                              size_t ws_size, hipStream_t stream) {
  const float* x      = (const float*)d_in[0];   // (1,64,16,16,16)
  const float* w_qkv  = (const float*)d_in[1];   // (192,64)
  const float* b_qkv  = (const float*)d_in[2];   // (192)
  const float* w_proj = (const float*)d_in[3];   // (64,64)
  const float* b_proj = (const float*)d_in[4];   // (64)
  float* out = (float*)d_out;

  char* ws = (char*)d_ws;
  float*          qkv = (float*)ws;                           // 192*4096 f32
  float*          ao  = (float*)(ws + 192 * NTOK * 4);        //  64*4096 f32
  unsigned short* vbf = (unsigned short*)(ws + 256 * NTOK * 4); // 8*16*4096 bf16

  qkv_kernel<<<dim3(256, 16), 256, 0, stream>>>(x, w_qkv, b_qkv, qkv, vbf);
  attn_kernel<<<dim3(256), 256, 0, stream>>>(qkv, vbf, ao);
  proj_kernel<<<dim3(64, 16), 256, 0, stream>>>(x, w_proj, b_proj, ao, out);
}
